// LSTMDecoder_6571299962966
// MI455X (gfx1250) — matrix operand order repaired and checked by reading, not compiled, not run on hardware
//
#include <hip/hip_runtime.h>
#include <math.h>

// ---------------- problem constants ----------------
#define VOCAB 32000
#define EMBED 512
#define HID   1024
#define SEQ   64
#define BSZ   32
#define SRC   64
#define NG    (4*HID)   // 4096 gate width

// ---------------- vector types for WMMA ----------------
typedef __attribute__((ext_vector_type(8)))  float  v8f;
typedef __attribute__((ext_vector_type(2)))  float  v2f;
typedef __attribute__((ext_vector_type(16))) __bf16 v16bf;

union B16Frag { uint4 u[2]; v16bf v; };

// ---------------------------------------------------------------------------
// Fragment loaders (layouts per ISA 7.12.2, wave32):
//  A frag lane L: row m0+(L&15), K = k0 + (L>=16?8:0) + {0..7} and +16..+23
//  B frag lane L: col n0+(L&15), K = k0 + (L>=16?8:0) + {0..7} and +16..+23
// ---------------------------------------------------------------------------
__device__ __forceinline__ v16bf load_afrag(const __bf16* __restrict__ A, int lda,
                                            int m0, int k0, int lane) {
  B16Frag a;
  const __bf16* ap = A + (size_t)(m0 + (lane & 15)) * lda + k0 + (lane >> 4) * 8;
  a.u[0] = *reinterpret_cast<const uint4*>(ap);
  a.u[1] = *reinterpret_cast<const uint4*>(ap + 16);
  return a.v;
}

__device__ __forceinline__ v16bf load_bfrag(const __bf16* __restrict__ W, int ldw,
                                            int n0, int k0, int lane) {
  B16Frag b;
  const __bf16* bp = W + (size_t)(n0 + (lane & 15)) * ldw + k0 + (lane >> 4) * 8;
  b.u[0] = *reinterpret_cast<const uint4*>(bp);
  b.u[1] = *reinterpret_cast<const uint4*>(bp + 16);
  return b.v;
}

// One k-chunk (K=32), both M-tiles of the fixed M=32 activation block share
// a single B (weight) fragment -> weight traffic read exactly once.
__device__ __forceinline__ void bf16_step2(const __bf16* __restrict__ A, int lda, int kA,
                                           const __bf16* __restrict__ W, int ldw,
                                           int n0, int kW, int lane,
                                           v8f& acc0, v8f& acc1) {
  v16bf b  = load_bfrag(W, ldw, n0, kW, lane);
  v16bf a0 = load_afrag(A, lda, 0,  kA, lane);
  v16bf a1 = load_afrag(A, lda, 16, kA, lane);
  acc0 = __builtin_amdgcn_wmma_f32_16x16x32_bf16(false, a0, false, b, (short)0, acc0, false, false);
  acc1 = __builtin_amdgcn_wmma_f32_16x16x32_bf16(false, a1, false, b, (short)0, acc1, false, false);
}

// ---------------------------------------------------------------------------
// bf16 GEMM, M fixed = 32:  C[32,N] = act( bias + [A1|A2] @ W^T + Ah @ Wh^T )
//   A1[32,K1], A2[32,K2] (K-concat), W[N,K1+K2], Ah[32,Kh], Wh[N,Kh].
//   One wave per 16-wide N tile, 2 accumulators cover M=32.
// ---------------------------------------------------------------------------
__global__ void gemm_bf16_kernel(const __bf16* __restrict__ A1, int K1,
                                 const __bf16* __restrict__ A2, int K2,
                                 const __bf16* __restrict__ W,
                                 const __bf16* __restrict__ Ah,
                                 const __bf16* __restrict__ Wh, int Kh,
                                 const float* __restrict__ bias,
                                 float* __restrict__ C,
                                 __bf16* __restrict__ Cbf,
                                 int N, int act) {
  const int wave = (blockIdx.x * blockDim.x + threadIdx.x) >> 5;
  const int lane = threadIdx.x & 31;
  if (wave >= (N >> 4)) return;
  const int n0 = wave << 4;
  const int idx = lane & 15;

  const float bv = bias ? bias[n0 + idx] : 0.0f;
  v8f acc0 = {bv, bv, bv, bv, bv, bv, bv, bv};
  v8f acc1 = acc0;

  const int ldw = K1 + K2;
  const __bf16* wrow = W ? W + (size_t)(n0 + idx) * ldw : nullptr;
  for (int k0 = 0; k0 < K1; k0 += 32) {
    __builtin_prefetch(wrow + k0 + 96, 0, 0);       // stream weights ahead
    bf16_step2(A1, K1, k0, W, ldw, n0, k0, lane, acc0, acc1);
  }
  for (int k0 = 0; k0 < K2; k0 += 32) {
    __builtin_prefetch(wrow + K1 + k0 + 96, 0, 0);
    bf16_step2(A2, K2, k0, W, ldw, n0, K1 + k0, lane, acc0, acc1);
  }
  const __bf16* whrow = Wh ? Wh + (size_t)(n0 + idx) * Kh : nullptr;
  for (int k0 = 0; k0 < Kh; k0 += 32) {
    __builtin_prefetch(whrow + k0 + 96, 0, 0);
    bf16_step2(Ah, Kh, k0, Wh, Kh, n0, k0, lane, acc0, acc1);
  }

  const int rbase = (lane >> 4) * 8;   // ISA C/D layout
  const int cc = n0 + idx;
#pragma unroll
  for (int i = 0; i < 8; ++i) {
    float v0 = acc0[i], v1 = acc1[i];
    if (act) { v0 = tanhf(v0); v1 = tanhf(v1); }
    C[(size_t)(rbase + i) * N + cc] = v0;
    C[(size_t)(16 + rbase + i) * N + cc] = v1;
    if (Cbf) {
      Cbf[(size_t)(rbase + i) * N + cc] = (__bf16)v0;
      Cbf[(size_t)(16 + rbase + i) * N + cc] = (__bf16)v1;
    }
  }
}

// ---------------------------------------------------------------------------
// Exact fp32 GEMM via V_WMMA_F32_16X16X4_F32 with 2x2 register blocking:
//   C[M,N] = A[M,K] @ W[N,K]^T + bias ;  M,N multiples of 32.
//   Each wave owns a 32x32 super-tile: 4 accumulators, 2 A + 2 B frags/step.
// ---------------------------------------------------------------------------
__global__ void gemm_f32_kernel(const float* __restrict__ A,
                                const float* __restrict__ W,
                                const float* __restrict__ bias,
                                float* __restrict__ C,
                                int M, int N, int K) {
  const int wave = (blockIdx.x * blockDim.x + threadIdx.x) >> 5;
  const int lane = threadIdx.x & 31;
  const int msup = M >> 5, nsup = N >> 5;
  if (wave >= msup * nsup) return;
  const int sm = wave % msup, sn = wave / msup;
  const int m0 = sm << 5, n0 = sn << 5;
  const int idx = lane & 15;
  const int kb = (lane >> 4) * 2;        // lanes 0-15: K {0,1}; 16-31: K {2,3}

  const float bv0 = bias ? bias[n0 + idx] : 0.0f;
  const float bv1 = bias ? bias[n0 + 16 + idx] : 0.0f;
  v8f acc00 = {bv0, bv0, bv0, bv0, bv0, bv0, bv0, bv0};
  v8f acc01 = {bv1, bv1, bv1, bv1, bv1, bv1, bv1, bv1};
  v8f acc10 = acc00, acc11 = acc01;

  const float* ap0 = A + (size_t)(m0 + idx) * K + kb;
  const float* ap1 = ap0 + (size_t)16 * K;
  const float* bp0 = W + (size_t)(n0 + idx) * K + kb;
  const float* bp1 = bp0 + (size_t)16 * K;
  for (int k0 = 0; k0 < K; k0 += 4) {
    __builtin_prefetch(bp0 + k0 + 32, 0, 0);
    __builtin_prefetch(bp1 + k0 + 32, 0, 0);
    v2f a0 = *reinterpret_cast<const v2f*>(ap0 + k0);
    v2f a1 = *reinterpret_cast<const v2f*>(ap1 + k0);
    v2f b0 = *reinterpret_cast<const v2f*>(bp0 + k0);
    v2f b1 = *reinterpret_cast<const v2f*>(bp1 + k0);
    acc00 = __builtin_amdgcn_wmma_f32_16x16x4_f32(false, a0, false, b0, (short)0, acc00, false, false);
    acc01 = __builtin_amdgcn_wmma_f32_16x16x4_f32(false, a0, false, b1, (short)0, acc01, false, false);
    acc10 = __builtin_amdgcn_wmma_f32_16x16x4_f32(false, a1, false, b0, (short)0, acc10, false, false);
    acc11 = __builtin_amdgcn_wmma_f32_16x16x4_f32(false, a1, false, b1, (short)0, acc11, false, false);
  }

  const int rbase = (lane >> 4) * 8;
  const int cc0 = n0 + idx, cc1 = n0 + 16 + idx;
#pragma unroll
  for (int i = 0; i < 8; ++i) {
    C[(size_t)(m0 + rbase + i) * N + cc0]      = acc00[i];
    C[(size_t)(m0 + rbase + i) * N + cc1]      = acc01[i];
    C[(size_t)(m0 + 16 + rbase + i) * N + cc0] = acc10[i];
    C[(size_t)(m0 + 16 + rbase + i) * N + cc1] = acc11[i];
  }
}

// ---------------- small elementwise / attention kernels ----------------
__global__ void f32_to_bf16_kernel(const float* __restrict__ s,
                                   __bf16* __restrict__ d, int n) {
  int i = blockIdx.x * blockDim.x + threadIdx.x;
  if (i < n) d[i] = (__bf16)s[i];
}

__global__ void add2_kernel(const float* __restrict__ a, const float* __restrict__ b,
                            float* __restrict__ o, int n) {
  int i = blockIdx.x * blockDim.x + threadIdx.x;
  if (i < n) o[i] = a[i] + b[i];
}

__global__ void fill_bf16_kernel(__bf16* __restrict__ p, int n) {
  int i = blockIdx.x * blockDim.x + threadIdx.x;
  if (i < n) p[i] = (__bf16)0.0f;
}

__global__ void copy_state_kernel(const float* __restrict__ s, float* __restrict__ d,
                                  __bf16* __restrict__ db, int n) {
  int i = blockIdx.x * blockDim.x + threadIdx.x;
  if (i < n) {
    float v = s[i];
    d[i] = v;
    if (db) db[i] = (__bf16)v;
  }
}

__global__ void embed_lookup_kernel(const int* __restrict__ y,
                                    const float* __restrict__ emb,
                                    __bf16* __restrict__ x, int n) {
  int i = blockIdx.x * blockDim.x + threadIdx.x;
  if (i >= n) return;
  int sb = i >> 9;          // EMBED = 512
  int e  = i & 511;
  int tok = y[sb];
  x[i] = (__bf16)emb[(size_t)tok * EMBED + e];
}

// gates [B,4H] f32 (order i,f,g,o) -> update c, h (f32) + h bf16 mirror
__global__ void lstm_cell_kernel(const float* __restrict__ g,
                                 float* __restrict__ c, float* __restrict__ h,
                                 __bf16* __restrict__ hb) {
  int i = blockIdx.x * blockDim.x + threadIdx.x;
  if (i >= BSZ * HID) return;
  int b = i >> 10, j = i & (HID - 1);
  const float* gr = g + (size_t)b * NG;
  float gi = gr[j], gf = gr[HID + j], gg = gr[2 * HID + j], go = gr[3 * HID + j];
  float si = 1.0f / (1.0f + expf(-gi));
  float sf = 1.0f / (1.0f + expf(-gf));
  float cv = sf * c[i] + si * tanhf(gg);
  c[i] = cv;
  float so = 1.0f / (1.0f + expf(-go));
  float hv = so * tanhf(cv);
  h[i] = hv;
  hb[i] = (__bf16)hv;
}

// scores[s][b] = dot(enc[s][b][:], q[b][:]) ; masked -> -inf. One wave per (s,b).
__global__ void attn_scores_kernel(const float* __restrict__ enc,
                                   const float* __restrict__ q,
                                   const unsigned char* __restrict__ mask,
                                   float* __restrict__ scores) {
  int wave = (blockIdx.x * blockDim.x + threadIdx.x) >> 5;
  int lane = threadIdx.x & 31;
  if (wave >= SRC * BSZ) return;
  int s = wave >> 5, b = wave & 31;     // BSZ = 32
  const float* ep = enc + (size_t)(s * BSZ + b) * HID;
  const float* qp = q + (size_t)b * HID;
  float sum = 0.0f;
  for (int hh = lane; hh < HID; hh += 32) sum += ep[hh] * qp[hh];
#pragma unroll
  for (int off = 16; off > 0; off >>= 1) sum += __shfl_xor(sum, off, 32);
  if (lane == 0)
    scores[s * BSZ + b] = mask[b * SRC + s] ? -__builtin_inff() : sum;
}

// softmax over s (=64) per batch column b; grid = BSZ, block = SRC
__global__ void softmax_kernel(const float* __restrict__ scores,
                               float* __restrict__ a) {
  __shared__ float red[SRC];
  int b = blockIdx.x, t = threadIdx.x;
  float v = scores[t * BSZ + b];
  red[t] = v;
  __syncthreads();
  for (int s = SRC / 2; s > 0; s >>= 1) {
    if (t < s) red[t] = fmaxf(red[t], red[t + s]);
    __syncthreads();
  }
  float m = red[0];
  __syncthreads();
  float e = expf(v - m);
  red[t] = e;
  __syncthreads();
  for (int s = SRC / 2; s > 0; s >>= 1) {
    if (t < s) red[t] += red[t + s];
    __syncthreads();
  }
  a[t * BSZ + b] = e / red[0];
}

// ctx[b][h] = sum_s a[s][b] * enc[s][b][h]  (bf16 out, feeds attn-out GEMM)
__global__ void attn_ctx_kernel(const float* __restrict__ a,
                                const float* __restrict__ enc,
                                __bf16* __restrict__ ctxb) {
  int i = blockIdx.x * blockDim.x + threadIdx.x;
  if (i >= BSZ * HID) return;
  int b = i >> 10, hh = i & (HID - 1);
  float sum = 0.0f;
  for (int s = 0; s < SRC; ++s)
    sum += a[s * BSZ + b] * enc[(size_t)(s * BSZ + b) * HID + hh];
  ctxb[i] = (__bf16)sum;
}

// ---------------------------------------------------------------------------
extern "C" void kernel_launch(void* const* d_in, const int* in_sizes, int n_in,
                              void* d_out, int out_size, void* d_ws, size_t ws_size,
                              hipStream_t stream) {
  const int*   y      = (const int*)d_in[0];
  const float* enc    = (const float*)d_in[1];
  const float* ehid   = (const float*)d_in[2];
  const float* ecell  = (const float*)d_in[3];
  const unsigned char* mask = (const unsigned char*)d_in[4];
  const float* emb    = (const float*)d_in[5];
  const float* w_ih0  = (const float*)d_in[6];
  const float* w_hh0  = (const float*)d_in[7];
  const float* b_ih0  = (const float*)d_in[8];
  const float* b_hh0  = (const float*)d_in[9];
  const float* w_ih1  = (const float*)d_in[10];
  const float* w_hh1  = (const float*)d_in[11];
  const float* b_ih1  = (const float*)d_in[12];
  const float* b_hh1  = (const float*)d_in[13];
  const float* attn_in_w  = (const float*)d_in[14];
  const float* attn_out_w = (const float*)d_in[15];
  const float* fc_w   = (const float*)d_in[16];
  const float* fc_b   = (const float*)d_in[17];
  float* out = (float*)d_out;

  char* ws = (char*)d_ws;
  size_t off = 0;
  auto alloc = [&](size_t bytes) -> void* {
    void* p = ws + off;
    off = (off + bytes + 255) & ~(size_t)255;
    return p;
  };

  __bf16* wb_ih0 = (__bf16*)alloc((size_t)NG * (EMBED + HID) * 2);
  __bf16* wb_hh0 = (__bf16*)alloc((size_t)NG * HID * 2);
  __bf16* wb_ih1 = (__bf16*)alloc((size_t)NG * HID * 2);
  __bf16* wb_hh1 = (__bf16*)alloc((size_t)NG * HID * 2);
  __bf16* wb_ai  = (__bf16*)alloc((size_t)HID * HID * 2);
  __bf16* wb_ao  = (__bf16*)alloc((size_t)HID * 2 * HID * 2);
  __bf16* xb     = (__bf16*)alloc((size_t)SEQ * BSZ * EMBED * 2);
  float*  bias0  = (float*)alloc(NG * 4);
  float*  bias1  = (float*)alloc(NG * 4);
  float*  h0     = (float*)alloc(BSZ * HID * 4);
  float*  c0     = (float*)alloc(BSZ * HID * 4);
  float*  h1     = (float*)alloc(BSZ * HID * 4);
  float*  c1     = (float*)alloc(BSZ * HID * 4);
  __bf16* h0b    = (__bf16*)alloc(BSZ * HID * 2);
  __bf16* h1b    = (__bf16*)alloc(BSZ * HID * 2);
  __bf16* feedb  = (__bf16*)alloc(BSZ * HID * 2);
  __bf16* ctxb   = (__bf16*)alloc(BSZ * HID * 2);
  float*  gates  = (float*)alloc((size_t)BSZ * NG * 4);
  float*  q      = (float*)alloc(BSZ * HID * 4);
  float*  scores = (float*)alloc(SRC * BSZ * 4);
  float*  attw   = (float*)alloc(SRC * BSZ * 4);
  float*  outs   = (float*)alloc((size_t)SEQ * BSZ * HID * 4);
  float*  proj   = (float*)alloc((size_t)SEQ * BSZ * EMBED * 4);
  (void)ws_size; (void)in_sizes; (void)n_in; (void)out_size;

  const int T = 256;
  auto nb = [](int n) { return (n + 255) / 256; };
  auto wb = [](int waves) { return (waves + 7) / 8; };   // 8 waves / block

  // ---- one-time (per call) weight conversions & setup ----
  f32_to_bf16_kernel<<<nb(NG * (EMBED + HID)), T, 0, stream>>>(w_ih0, wb_ih0, NG * (EMBED + HID));
  f32_to_bf16_kernel<<<nb(NG * HID), T, 0, stream>>>(w_hh0, wb_hh0, NG * HID);
  f32_to_bf16_kernel<<<nb(NG * HID), T, 0, stream>>>(w_ih1, wb_ih1, NG * HID);
  f32_to_bf16_kernel<<<nb(NG * HID), T, 0, stream>>>(w_hh1, wb_hh1, NG * HID);
  f32_to_bf16_kernel<<<nb(HID * HID), T, 0, stream>>>(attn_in_w, wb_ai, HID * HID);
  f32_to_bf16_kernel<<<nb(HID * 2 * HID), T, 0, stream>>>(attn_out_w, wb_ao, HID * 2 * HID);
  add2_kernel<<<nb(NG), T, 0, stream>>>(b_ih0, b_hh0, bias0, NG);
  add2_kernel<<<nb(NG), T, 0, stream>>>(b_ih1, b_hh1, bias1, NG);
  embed_lookup_kernel<<<nb(SEQ * BSZ * EMBED), T, 0, stream>>>(y, emb, xb, SEQ * BSZ * EMBED);
  copy_state_kernel<<<nb(BSZ * HID), T, 0, stream>>>(ehid,              h0, h0b, BSZ * HID);
  copy_state_kernel<<<nb(BSZ * HID), T, 0, stream>>>(ehid + BSZ * HID,  h1, h1b, BSZ * HID);
  copy_state_kernel<<<nb(BSZ * HID), T, 0, stream>>>(ecell,             c0, (__bf16*)nullptr, BSZ * HID);
  copy_state_kernel<<<nb(BSZ * HID), T, 0, stream>>>(ecell + BSZ * HID, c1, (__bf16*)nullptr, BSZ * HID);
  fill_bf16_kernel<<<nb(BSZ * HID), T, 0, stream>>>(feedb, BSZ * HID);

  const int gate_waves = NG / 16;    // 256 (each wave covers full M=32)
  const int hh_waves   = HID / 16;   // 64

  // ---- sequential decode loop ----
  for (int t = 0; t < SEQ; ++t) {
    const __bf16* xt = xb + (size_t)t * BSZ * EMBED;
    // layer 0 gates: bias0 + [x_t | feed] @ w_ih0^T + h0 @ w_hh0^T
    gemm_bf16_kernel<<<wb(gate_waves), T, 0, stream>>>(
        xt, EMBED, feedb, HID, wb_ih0, h0b, wb_hh0, HID,
        bias0, gates, (__bf16*)nullptr, NG, 0);
    lstm_cell_kernel<<<nb(BSZ * HID), T, 0, stream>>>(gates, c0, h0, h0b);
    // layer 1 gates: bias1 + h0 @ w_ih1^T + h1 @ w_hh1^T
    gemm_bf16_kernel<<<wb(gate_waves), T, 0, stream>>>(
        h0b, HID, (const __bf16*)nullptr, 0, wb_ih1, h1b, wb_hh1, HID,
        bias1, gates, (__bf16*)nullptr, NG, 0);
    lstm_cell_kernel<<<nb(BSZ * HID), T, 0, stream>>>(gates, c1, h1, h1b);
    // attention: q = h1 @ attn_in_w^T
    gemm_bf16_kernel<<<wb(hh_waves), T, 0, stream>>>(
        h1b, HID, (const __bf16*)nullptr, 0, wb_ai,
        (const __bf16*)nullptr, (const __bf16*)nullptr, 0,
        (const float*)nullptr, q, (__bf16*)nullptr, HID, 0);
    attn_scores_kernel<<<(SRC * BSZ * 32) / T, T, 0, stream>>>(enc, q, mask, scores);
    softmax_kernel<<<BSZ, SRC, 0, stream>>>(scores, attw);
    attn_ctx_kernel<<<nb(BSZ * HID), T, 0, stream>>>(attw, enc, ctxb);
    // out = tanh([ctx | h1] @ attn_out_w^T) -> outs[t] (f32) and feed (bf16)
    gemm_bf16_kernel<<<wb(hh_waves), T, 0, stream>>>(
        ctxb, HID, h1b, HID, wb_ao,
        (const __bf16*)nullptr, (const __bf16*)nullptr, 0,
        (const float*)nullptr, outs + (size_t)t * BSZ * HID, feedb, HID, 1);
  }

  // ---- output projection in exact fp32 WMMA (2x2 blocked) ----
  const int fc_waves = (SEQ * BSZ / 32) * (EMBED / 32);   // 64*16
  gemm_f32_kernel<<<wb(fc_waves), T, 0, stream>>>(outs, fc_w, fc_b, proj,
                                                  SEQ * BSZ, EMBED, HID);
  const int lg_waves = (SEQ * BSZ / 32) * (VOCAB / 32);   // 64*1000
  gemm_f32_kernel<<<wb(lg_waves), T, 0, stream>>>(proj, emb, (const float*)nullptr,
                                                  out, SEQ * BSZ, VOCAB, EMBED);
}
